// LocalAttentionBlock_74594991997388
// MI455X (gfx1250) — matrix operand order repaired and checked by reading, not compiled, not run on hardware
//
#include <hip/hip_runtime.h>

typedef __bf16 v16bf __attribute__((ext_vector_type(16)));
typedef float  v8f   __attribute__((ext_vector_type(8)));
typedef float  v4f   __attribute__((ext_vector_type(4)));
typedef unsigned int v4u __attribute__((ext_vector_type(4)));
typedef unsigned int v2u __attribute__((ext_vector_type(2)));

union Frag16 { v16bf bf; v4u u[2]; };

#define C_T  2048
#define C_NH 10
#define C_HD 256
#define C_W  2560

// hard scheduling fence: no instruction may be moved across
#define SCHED_FENCE() __builtin_amdgcn_sched_barrier(0)

__device__ __forceinline__ unsigned short f2bf(float f) {
  unsigned int u = __float_as_uint(f);
  u += 0x7FFFu + ((u >> 16) & 1u);      // round-to-nearest-even
  return (unsigned short)(u >> 16);
}
__device__ __forceinline__ float bf2f(unsigned short h) {
  return __uint_as_float(((unsigned int)h) << 16);
}

// ---------------------------------------------------------------------------
// Elementwise f32 -> bf16 (vectorized x4).
// ---------------------------------------------------------------------------
__global__ __launch_bounds__(256) void cvt_kernel(
    const float* __restrict__ in, unsigned short* __restrict__ out, int n4) {
  const int i = blockIdx.x * 256 + threadIdx.x;
  if (i >= n4) return;
  v4f d = ((const v4f*)in)[i];
  v2u st;
  st.x = (unsigned)f2bf(d.x) | ((unsigned)f2bf(d.y) << 16);
  st.y = (unsigned)f2bf(d.z) | ((unsigned)f2bf(d.w) << 16);
  ((v2u*)out)[i] = st;
}

// ---------------------------------------------------------------------------
// NT GEMM, all-bf16: Out[m,n] = sum_k A[m,k]*W[n,k] (+bias[n])
// Block 256 thr = 8 waves (4m x 2n); block tile 128x64; k-step 32.
// Double-buffered LDS staged by GLOBAL_LOAD_ASYNC_TO_LDS_B128.
// In-order ASYNCcnt completion: with 3 transfers per buffer,
// "s_wait_asynccnt 3" after issuing the next buffer == current buffer landed.
// ---------------------------------------------------------------------------
template <bool BIASOUT>
__global__ __launch_bounds__(256) void gemm_bf16_kernel(
    const unsigned short* __restrict__ A, const unsigned short* __restrict__ Bw,
    const float* __restrict__ bias, void* __restrict__ Out,
    int M, int N, int K) {
  __shared__ __align__(16) unsigned short lA[2][8 * 512];   // 2 x 8KB
  __shared__ __align__(16) unsigned short lB[2][4 * 512];   // 2 x 4KB
  const int tid  = threadIdx.x;
  const int lane = tid & 31;
  const int w    = tid >> 5;
  const int wm   = w & 3;
  const int wn   = w >> 2;
  const int mBase = blockIdx.y * 128;
  const int nBase = blockIdx.x * 64;

  const v8f zero8 = {0.f, 0.f, 0.f, 0.f, 0.f, 0.f, 0.f, 0.f};
  v8f acc[2][2];
#pragma unroll
  for (int i = 0; i < 2; ++i)
#pragma unroll
    for (int j = 0; j < 2; ++j) acc[i][j] = zero8;

  // per-thread 16B-chunk coordinates: 8 consecutive k starting at k8
  const int k8  = (tid & 3) * 8;                 // 0,8,16,24
  const int hsA = (k8 >> 3) & 1;                 // A lane-half
  const int eA  = ((k8 >> 4) & 1) * 8;           // A element base
  const int hsB = (k8 >> 3) & 1;                 // B lane-half
  const int eB  = ((k8 >> 4) & 1) * 8;           // B element base (0 or 8)
  const int rowA = tid >> 2;                     // 0..63 (row / n index)

  const unsigned ldsA0 = (unsigned)(size_t)&lA[0][(rowA >> 4) * 512 +
                                                 ((rowA & 15) + hsA * 16) * 16 + eA];
  const unsigned ldsB0 = (unsigned)(size_t)&lB[0][(rowA >> 4) * 512 +
                                                 ((rowA & 15) + hsB * 16) * 16 + eB];
  const unsigned short* gA0 = A  + (size_t)(mBase + rowA) * K + k8;
  const unsigned short* gA1 = gA0 + (size_t)64 * K;
  const unsigned short* gB0 = Bw + (size_t)(nBase + rowA) * K + k8;

  const int kIters = K >> 5;

#define GEMM_ISSUE(kbn, buf)                                                   \
  {                                                                            \
    const unsigned la0 = ldsA0 + (buf) * 8192;                                 \
    const unsigned la1 = la0 + 4096;                                           \
    const unsigned lb0 = ldsB0 + (buf) * 4096;                                 \
    const unsigned short* pa0 = gA0 + (kbn) * 32;                              \
    const unsigned short* pa1 = gA1 + (kbn) * 32;                              \
    const unsigned short* pb0 = gB0 + (kbn) * 32;                              \
    asm volatile(                                                              \
        "global_load_async_to_lds_b128 %0, %3, off\n\t"                        \
        "global_load_async_to_lds_b128 %1, %4, off\n\t"                        \
        "global_load_async_to_lds_b128 %2, %5, off"                            \
        :: "v"(la0), "v"(la1), "v"(lb0), "v"(pa0), "v"(pa1), "v"(pb0)          \
        : "memory");                                                           \
  }

#define GEMM_COMPUTE(buf)                                                      \
  {                                                                            \
    Frag16 af[2], bfr[2];                                                      \
    _Pragma("unroll")                                                          \
    for (int i = 0; i < 2; ++i) {                                              \
      const v4u* p = (const v4u*)&lA[buf][(wm * 2 + i) * 512 + lane * 16];     \
      af[i].u[0] = p[0]; af[i].u[1] = p[1];                                    \
    }                                                                          \
    _Pragma("unroll")                                                          \
    for (int j = 0; j < 2; ++j) {                                              \
      const v4u* p = (const v4u*)&lB[buf][(wn * 2 + j) * 512 + lane * 16];     \
      bfr[j].u[0] = p[0]; bfr[j].u[1] = p[1];                                  \
    }                                                                          \
    SCHED_FENCE();                                                             \
    _Pragma("unroll")                                                          \
    for (int i = 0; i < 2; ++i)                                                \
      _Pragma("unroll")                                                        \
      for (int j = 0; j < 2; ++j)                                              \
        acc[i][j] = __builtin_amdgcn_wmma_f32_16x16x32_bf16(                   \
            false, af[i].bf, false, bfr[j].bf, (short)0, acc[i][j],            \
            false, false);                                                     \
  }

  GEMM_ISSUE(0, 0);
  for (int kb = 0; kb < kIters - 1; ++kb) {
    const int cur = kb & 1;
    GEMM_ISSUE(kb + 1, cur ^ 1);
    asm volatile("s_wait_asynccnt 3" ::: "memory");   // current buffer landed
    __syncthreads();
    GEMM_COMPUTE(cur);
    __syncthreads();
  }
  asm volatile("s_wait_asynccnt 0" ::: "memory");
  __syncthreads();
  {
    const int cur = (kIters - 1) & 1;
    GEMM_COMPUTE(cur);
  }

  // epilogue (C layout: lane holds col n=lane&15, rows (lane>>4)*8+r)
  const int hsL = lane >> 4, nn = lane & 15;
#pragma unroll
  for (int i = 0; i < 2; ++i)
#pragma unroll
    for (int j = 0; j < 2; ++j) {
      const int col = nBase + wn * 32 + j * 16 + nn;
      const float bv = BIASOUT ? bias[col] : 0.f;
#pragma unroll
      for (int r = 0; r < 8; ++r) {
        const int row = mBase + wm * 32 + i * 16 + hsL * 8 + r;
        const float v = acc[i][j][r] + bv;
        if (BIASOUT)
          ((float*)Out)[(size_t)row * N + col] = v;
        else
          ((unsigned short*)Out)[(size_t)row * N + col] = f2bf(v);
      }
    }
#undef GEMM_ISSUE
#undef GEMM_COMPUTE
}

// ---------------------------------------------------------------------------
// RoPE in-place on bf16 rows: head slice at blockIdx.y*256 of each row.
// ---------------------------------------------------------------------------
__global__ void rope_kernel(unsigned short* __restrict__ buf,
                            const int* __restrict__ segpos, int rowLen) {
  const int row = blockIdx.x;            // b*T + t
  const int i   = threadIdx.x;           // 0..63
  const size_t base = (size_t)row * rowLen + (size_t)blockIdx.y * 256;
  const float a  = bf2f(buf[base + i]);
  const float bq = bf2f(buf[base + 64 + i]);
  const float pos = (float)segpos[row];
  const float inv = __expf(-(float)i * 0.14391157f);   // 10000^(-i/64)
  const float th  = pos * inv;
  float s, c;
  __sincosf(th, &s, &c);
  buf[base + i]      = f2bf(a * c - bq * s);
  buf[base + 64 + i] = f2bf(bq * c + a * s);
}

// v_bf (B*T, 256) -> v_t (B, 256, T)
__global__ void vtrans_kernel(const unsigned short* __restrict__ v,
                              unsigned short* __restrict__ vt) {
  const int row = blockIdx.x;
  const int d   = threadIdx.x;
  const int b = row >> 11, t = row & (C_T - 1);
  vt[((size_t)b * C_HD + d) * C_T + t] = v[(size_t)row * C_HD + d];
}

// ---------------------------------------------------------------------------
// Flash attention (causal, MQA). Grid (T/128, NH, B), 256 thr = 8 waves.
// Wave w owns 16 q rows. sched_barrier(0) fences separate each group's
// fragment loads from its WMMAs: loads issue as one clause into distinct
// registers; next group's loads may slide between this group's WMMAs.
// ---------------------------------------------------------------------------
__global__ __launch_bounds__(256) void attn_kernel(
    const unsigned short* __restrict__ qbf, const unsigned short* __restrict__ kbf,
    const unsigned short* __restrict__ vt, unsigned short* __restrict__ enc) {
  __shared__ __align__(16) unsigned short lP[8 * 512];   // 1KB per wave
  const int lane = threadIdx.x & 31;
  const int w    = threadIdx.x >> 5;
  const int qb = blockIdx.x, h = blockIdx.y, b = blockIdx.z;
  const int m0  = qb * 128 + w * 16;
  const int hsL = lane >> 4, ln = lane & 15;

  const unsigned short* qrow =
      qbf + ((size_t)(b * C_T) + m0 + ln) * C_W + h * 256 + hsL * 16;

  const v8f zero8 = {0.f, 0.f, 0.f, 0.f, 0.f, 0.f, 0.f, 0.f};
  v8f o[16];
#pragma unroll
  for (int n = 0; n < 16; ++n) o[n] = zero8;
  float mrun[8], lrun[8];
#pragma unroll
  for (int r = 0; r < 8; ++r) { mrun[r] = -3.0e38f; lrun[r] = 0.f; }

  unsigned short* pbase = &lP[w * 512];
  const int sEnd = m0 + 15;

  for (int s0 = 0; s0 <= sEnd; s0 += 32) {
    // ---- logits = Q K^T: 4 groups of (2 kd) x (2 n-tiles) ----
    v8f lg0 = zero8, lg1 = zero8;
    const unsigned short* kp0 =
        kbf + ((size_t)(b * C_T) + s0 + ln) * C_HD + hsL * 16;
    const unsigned short* kp1 = kp0 + 16 * C_HD;
#pragma unroll
    for (int g = 0; g < 4; ++g) {
      Frag16 qg[2], b0[2], b1[2];
#pragma unroll
      for (int q = 0; q < 2; ++q) {
        const int kd = g * 2 + q;
        qg[q].u[0] = *(const v4u*)(qrow + kd * 32);
        qg[q].u[1] = *(const v4u*)(qrow + kd * 32 + 8);
        b0[q].u[0] = *(const v4u*)(kp0 + kd * 32);
        b0[q].u[1] = *(const v4u*)(kp0 + kd * 32 + 8);
        b1[q].u[0] = *(const v4u*)(kp1 + kd * 32);
        b1[q].u[1] = *(const v4u*)(kp1 + kd * 32 + 8);
      }
      SCHED_FENCE();   // loads above issue as a clause; WMMAs stay below
      lg0 = __builtin_amdgcn_wmma_f32_16x16x32_bf16(
          false, qg[0].bf, false, b0[0].bf, (short)0, lg0, false, false);
      lg1 = __builtin_amdgcn_wmma_f32_16x16x32_bf16(
          false, qg[0].bf, false, b1[0].bf, (short)0, lg1, false, false);
      lg0 = __builtin_amdgcn_wmma_f32_16x16x32_bf16(
          false, qg[1].bf, false, b0[1].bf, (short)0, lg0, false, false);
      lg1 = __builtin_amdgcn_wmma_f32_16x16x32_bf16(
          false, qg[1].bf, false, b1[1].bf, (short)0, lg1, false, false);
    }
    // ---- scale + causal mask + online softmax ----
    float p0[8], p1[8], sc[8];
#pragma unroll
    for (int r = 0; r < 8; ++r) {
      const int mrow = m0 + hsL * 8 + r;
      float a = lg0[r] * 0.0625f;
      float c = lg1[r] * 0.0625f;
      if (s0 + ln      > mrow) a = -3.0e38f;
      if (s0 + 16 + ln > mrow) c = -3.0e38f;
      float t = fmaxf(a, c);
      t = fmaxf(t, __shfl_xor(t, 1, 32));
      t = fmaxf(t, __shfl_xor(t, 2, 32));
      t = fmaxf(t, __shfl_xor(t, 4, 32));
      t = fmaxf(t, __shfl_xor(t, 8, 32));
      const float mn    = fmaxf(mrun[r], t);
      const float scale = __expf(mrun[r] - mn);
      const float ea = __expf(a - mn);
      const float ec = __expf(c - mn);
      float rs = ea + ec;
      rs += __shfl_xor(rs, 1, 32);
      rs += __shfl_xor(rs, 2, 32);
      rs += __shfl_xor(rs, 4, 32);
      rs += __shfl_xor(rs, 8, 32);
      lrun[r] = lrun[r] * scale + rs;
      mrun[r] = mn;
      sc[r] = scale; p0[r] = ea; p1[r] = ec;
    }
#pragma unroll
    for (int n = 0; n < 16; ++n)
#pragma unroll
      for (int r = 0; r < 8; ++r) o[n][r] *= sc[r];

    // ---- P: C layout -> A layout via per-wave LDS (in-wave DS ordering) ----
#pragma unroll
    for (int r = 0; r < 8; ++r) {
      pbase[(hsL * 8 + r) * 32 + ln]      = f2bf(p0[r]);
      pbase[(hsL * 8 + r) * 32 + 16 + ln] = f2bf(p1[r]);
    }
    asm volatile("s_wait_dscnt 0" ::: "memory");
    Frag16 pf;
    pf.u[0] = *(const v4u*)&pbase[ln * 32 + hsL * 16];
    pf.u[1] = *(const v4u*)&pbase[ln * 32 + hsL * 16 + 8];

    // ---- O += P V: pairs with independent accumulator chains ----
    const unsigned short* vb =
        vt + ((size_t)b * C_HD + ln) * C_T + s0 + hsL * 16;
#pragma unroll
    for (int g = 0; g < 8; ++g) {
      Frag16 vf[2];
#pragma unroll
      for (int q = 0; q < 2; ++q) {
        const unsigned short* vp = vb + (size_t)((g * 2 + q) * 16) * C_T;
        vf[q].u[0] = *(const v4u*)(vp);
        vf[q].u[1] = *(const v4u*)(vp + 8);
      }
      SCHED_FENCE();
      o[g * 2]     = __builtin_amdgcn_wmma_f32_16x16x32_bf16(
          false, pf.bf, false, vf[0].bf, (short)0, o[g * 2], false, false);
      o[g * 2 + 1] = __builtin_amdgcn_wmma_f32_16x16x32_bf16(
          false, pf.bf, false, vf[1].bf, (short)0, o[g * 2 + 1], false, false);
    }
  }

  // ---- normalize + store encoded (bf16 feeds final GEMM) ----
  unsigned short* op =
      enc + ((size_t)(b * C_T) + m0 + hsL * 8) * C_W + h * 256 + ln;
#pragma unroll
  for (int r = 0; r < 8; ++r) {
    const float inv = 1.0f / lrun[r];
#pragma unroll
    for (int n = 0; n < 16; ++n)
      op[(size_t)r * C_W + n * 16] = f2bf(o[n][r] * inv);
  }
}

// ---------------------------------------------------------------------------
extern "C" void kernel_launch(void* const* d_in, const int* in_sizes, int n_in,
                              void* d_out, int out_size, void* d_ws, size_t ws_size,
                              hipStream_t stream) {
  (void)in_sizes; (void)n_in; (void)out_size; (void)ws_size;
  const float* x   = (const float*)d_in[0];
  const int*   sp  = (const int*)d_in[1];
  const float* wq  = (const float*)d_in[2];
  const float* wk  = (const float*)d_in[3];
  const float* wv  = (const float*)d_in[4];
  const float* wf  = (const float*)d_in[5];
  const float* bfi = (const float*)d_in[6];

  const int Bb = 2, T = C_T, W = C_W, NH = C_NH, HD = C_HD;
  const int M = Bb * T;

  unsigned short* ws    = (unsigned short*)d_ws;
  unsigned short* x_bf  = ws;                           // M*W
  unsigned short* q_bf  = x_bf  + (size_t)M * W;        // M*W
  unsigned short* k_bf  = q_bf  + (size_t)M * W;        // M*HD
  unsigned short* v_bf  = k_bf  + (size_t)M * HD;       // M*HD
  unsigned short* v_t   = v_bf  + (size_t)M * HD;       // Bb*HD*T
  unsigned short* enc   = v_t   + (size_t)M * HD;       // M*W
  unsigned short* wq_bf = enc   + (size_t)M * W;        // W*W
  unsigned short* wk_bf = wq_bf + (size_t)W * W;        // HD*W
  unsigned short* wv_bf = wk_bf + (size_t)HD * W;       // HD*W
  unsigned short* wf_bf = wv_bf + (size_t)HD * W;       // W*W

  const dim3 blk(256);
  // one-pass f32 -> bf16 conversions
  cvt_kernel<<<dim3((M * W) / 1024),  blk, 0, stream>>>(x,  x_bf,  (M * W) / 4);
  cvt_kernel<<<dim3((W * W) / 1024),  blk, 0, stream>>>(wq, wq_bf, (W * W) / 4);
  cvt_kernel<<<dim3((HD * W) / 1024), blk, 0, stream>>>(wk, wk_bf, (HD * W) / 4);
  cvt_kernel<<<dim3((HD * W) / 1024), blk, 0, stream>>>(wv, wv_bf, (HD * W) / 4);
  cvt_kernel<<<dim3((W * W) / 1024),  blk, 0, stream>>>(wf, wf_bf, (W * W) / 4);
  // Q/K/V projections (double-buffered async-to-LDS WMMA GEMMs)
  gemm_bf16_kernel<false><<<dim3(W / 64,  M / 128), blk, 0, stream>>>(
      x_bf, wq_bf, nullptr, q_bf, M, W, W);
  gemm_bf16_kernel<false><<<dim3(HD / 64, M / 128), blk, 0, stream>>>(
      x_bf, wk_bf, nullptr, k_bf, M, HD, W);
  gemm_bf16_kernel<false><<<dim3(HD / 64, M / 128), blk, 0, stream>>>(
      x_bf, wv_bf, nullptr, v_bf, M, HD, W);
  // RoPE on Q (10 heads) and K (1 head)
  rope_kernel<<<dim3(M, NH), dim3(64), 0, stream>>>(q_bf, sp, W);
  rope_kernel<<<dim3(M, 1),  dim3(64), 0, stream>>>(k_bf, sp, HD);
  // V transpose for contiguous PV B-fragments
  vtrans_kernel<<<dim3(M), dim3(256), 0, stream>>>(v_bf, v_t);
  // causal flash attention
  attn_kernel<<<dim3(T / 128, NH, Bb), blk, 0, stream>>>(q_bf, k_bf, v_t, enc);
  // final projection (bf16 in, f32 + bias out)
  gemm_bf16_kernel<true><<<dim3(W / 64, M / 128), blk, 0, stream>>>(
      enc, wf_bf, bfi, (float*)d_out, M, W, W);
}
